// TorchLSTMNet_85676007620689
// MI455X (gfx1250) — compile-verified, matrix operand order checked by reading, not run on hardware
//
#include <hip/hip_runtime.h>

// ---------------- types ----------------
typedef __attribute__((ext_vector_type(16))) __bf16 v16bf;
typedef __attribute__((ext_vector_type(8)))  float  v8f;
typedef __attribute__((ext_vector_type(4)))  unsigned int u32x4;
typedef __attribute__((ext_vector_type(8)))  int i32x8;
typedef __attribute__((ext_vector_type(4)))  int i32x4;

union FragA { v16bf v; u32x4 q[2]; };   // one 16x32 bf16 A/B operand (8 VGPRs)
union AccU  { v8f  v; float f[8]; };    // one 16x16 f32 C/D operand (8 VGPRs)

#define LSTM_B 64
#define LSTM_T 2048
#define LSTM_I 64
#define LSTM_H 256
#define LSTM_G 1024   // 4*H

// ---------------- prep kernels ----------------
__global__ void cvt_bf16_kernel(const float* __restrict__ src,
                                __bf16* __restrict__ dst, int n) {
  int i = blockIdx.x * blockDim.x + threadIdx.x;
  if (i < n) dst[i] = (__bf16)src[i];
}

__global__ void bias_sum_kernel(const float* __restrict__ a,
                                const float* __restrict__ b,
                                float* __restrict__ o) {
  int i = blockIdx.x * blockDim.x + threadIdx.x;
  if (i < LSTM_G) o[i] = a[i] + b[i];
}

// ---------------- TDM helper: 2D tile load, data_size = 4B ----------------
// D# per CDNA5 ISA ch.8: group0 = {count/flags, lds_addr, global_addr, type=2},
// group1 = {data_size, tensor_dim0/1, tile_dim0/1, tensor_dim0_stride}.
__device__ __forceinline__ void tdm_load_2d(unsigned lds_off,
                                            unsigned long long gaddr,
                                            unsigned tile_w_dw, unsigned tile_h,
                                            unsigned stride_dw) {
#if __has_builtin(__builtin_amdgcn_tensor_load_to_lds)
  u32x4 g0;
  g0[0] = 1u;                                   // count=1, no gather
  g0[1] = lds_off;                              // LDS byte address
  g0[2] = (unsigned)gaddr;                      // global addr [31:0]
  g0[3] = (unsigned)(gaddr >> 32) | 0x80000000u;// addr [56:32] | type=2<<30
  i32x8 g1;
  g1[0] = (int)(2u << 16);                      // data_size = 2 (4 bytes)
  g1[1] = (int)(tile_w_dw << 16);               // tensor_dim0 [15:0]
  g1[2] = (int)((tile_w_dw >> 16) | (tile_h << 16));  // dim0 hi | tensor_dim1 lo
  g1[3] = (int)((tile_h >> 16) | (tile_w_dw << 16));  // dim1 hi | tile_dim0
  g1[4] = (int)tile_h;                          // tile_dim1 (tile_dim2 = 0)
  g1[5] = (int)stride_dw;                       // tensor_dim0_stride [31:0]
  g1[6] = 0;
  g1[7] = 0;
  i32x4 z4 = {0, 0, 0, 0};
#if defined(__clang_major__) && __clang_major__ >= 23
  i32x8 z8 = {0, 0, 0, 0, 0, 0, 0, 0};
  __builtin_amdgcn_tensor_load_to_lds(g0, g1, z4, z4, z8, 0);
#else
  __builtin_amdgcn_tensor_load_to_lds(g0, g1, z4, z4, 0);
#endif
#endif
}

// ---------------- persistent LSTM kernel ----------------
// grid = 4 WGs x 512 threads (16 waves, wave32). Batch tile of 16 per WG.
// Wave w owns hidden range j in [16w,16w+16) for ALL FOUR gates (i,f,g,o):
// c/h update is register-local to the wave; no gate LDS round-trip.
// W_hh (bf16, 512 KB, L2-resident) is streamed per step through a 2x64KB
// LDS ping-pong via TENSOR_LOAD_TO_LDS, one 64KB K-slice (32 of 256 K) ahead.
__launch_bounds__(512, 1)
__global__ void lstm_persist_kernel(const __bf16* __restrict__ Whh,   // [1024,256] bf16
                                    const __bf16* __restrict__ Wih,   // [1024,64]  bf16
                                    const __bf16* __restrict__ Xbf,   // [64,2048,64] bf16
                                    const float*  __restrict__ Bsum,  // [1024] b_ih+b_hh
                                    const float*  __restrict__ Wo,    // [256]
                                    const float*  __restrict__ Bo,    // [1]
                                    float*        __restrict__ Out) { // [64,2048]
  __shared__ __align__(16) __bf16 wbuf[2][LSTM_G * 32]; // 2 x 64KB K-slice ping-pong
  __shared__ __align__(16) __bf16 hsh[16 * LSTM_H];     // 8KB h in WMMA-A layout
  __shared__ float outacc[16];

  const int tid  = threadIdx.x;
  const int lane = tid & 31;
  const int wv   = tid >> 5;            // 0..15 : hidden range [16wv,16wv+16)
  const int bm   = blockIdx.x;          // batch tile rows bm*16..bm*16+15
  const int l15  = lane & 15;
  const int hiK  = (lane < 16) ? 0 : 8; // ISA 7.12.2 lane K-offset (== M offset for C/D)

  const bool haveTDM =
#if __has_builtin(__builtin_amdgcn_tensor_load_to_lds)
      true;
#else
      false;
#endif

  // ---- one-time init ----
  for (int i = tid; i < 16 * LSTM_H; i += 512) hsh[i] = (__bf16)0.0f;
  if (tid < 16) outacc[tid] = 0.0f;

  // W_ih fragments register-resident: 4 gate tiles x 2 K-chunks (64 VGPRs).
  FragA bih[4][2];
  float bias4[4];
#pragma unroll
  for (int gi = 0; gi < 4; ++gi) {
    const int n = gi * LSTM_H + wv * 16 + l15;  // gate-output row this lane feeds
    bias4[gi] = Bsum[n];
    const __bf16* wr = Wih + (size_t)n * LSTM_I;
#pragma unroll
    for (int kc = 0; kc < 2; ++kc) {
      const int kb = kc * 32 + hiK;
      bih[gi][kc].q[0] = *(const u32x4*)(wr + kb);
      bih[gi][kc].q[1] = *(const u32x4*)(wr + kb + 16);
    }
  }
  const float wo = Wo[wv * 16 + l15];   // output-projection weight for this lane's j
  const float bo = Bo[0];
  float creg[8];                        // c[m=r+hiK][j] lives in the wave forever
#pragma unroll
  for (int r = 0; r < 8; ++r) creg[r] = 0.f;

  const unsigned long long whh_base = (unsigned long long)Whh;
  const unsigned buf_lds[2] = {(unsigned)(size_t)&wbuf[0][0],
                               (unsigned)(size_t)&wbuf[1][0]};

  // Kick off stage 0 (K columns [0,32) of W_hh, 64KB).
  if (haveTDM && tid == 0)
    tdm_load_2d(buf_lds[0], whh_base, /*tile_w_dw=*/16, /*tile_h=*/LSTM_G,
                /*stride_dw=*/128);

  // Per-lane A-row base for x: M = l15 -> batch bm*16+l15
  const __bf16* xrow = Xbf + ((size_t)(bm * 16 + l15) * LSTM_T) * LSTM_I;

  __syncthreads();

  for (int t = 0; t < LSTM_T; ++t) {
    // ---- (A) preload h A-fragments (8 K-chunks) and x A-fragments ----
    FragA ah[8];
#pragma unroll
    for (int kc = 0; kc < 8; ++kc) {
      const __bf16* hp = &hsh[l15 * LSTM_H + kc * 32 + hiK];
      ah[kc].q[0] = *(const u32x4*)hp;
      ah[kc].q[1] = *(const u32x4*)(hp + 16);
    }
    const __bf16* xp = xrow + (size_t)t * LSTM_I;
    FragA ax[2];
#pragma unroll
    for (int kc = 0; kc < 2; ++kc) {
      const int kb = kc * 32 + hiK;
      ax[kc].q[0] = *(const u32x4*)(xp + kb);
      ax[kc].q[1] = *(const u32x4*)(xp + kb + 16);
    }
    if (t + 1 < LSTM_T) __builtin_prefetch(xp + LSTM_I, 0, 1);  // global_prefetch_b8
    __syncthreads();  // all hsh reads done before this step's h writes

    // ---- (B) gates = bias + x@Wih^T  (K=64, W_ih in registers) ----
    AccU acc[4];
#pragma unroll
    for (int gi = 0; gi < 4; ++gi) {
#pragma unroll
      for (int r = 0; r < 8; ++r) acc[gi].f[r] = bias4[gi];
#pragma unroll
      for (int kc = 0; kc < 2; ++kc)
        acc[gi].v = __builtin_amdgcn_wmma_f32_16x16x32_bf16(
            false, ax[kc].v, false, bih[gi][kc].v, (short)0, acc[gi].v, false, false);
    }

    // ---- (C) += h@Whh^T, streaming K-slices through LDS ping-pong ----
    for (int s = 0; s < 8; ++s) {
      if (haveTDM && wv == 0) __builtin_amdgcn_s_wait_tensorcnt(0);  // slice s landed
      __syncthreads();  // also: everyone finished reading slice s-1's buffer
      const int ns = (s + 1) & 7;  // next slice (wraps into next timestep)
      if (haveTDM) {
        if (tid == 0)
          tdm_load_2d(buf_lds[(s + 1) & 1], whh_base + (unsigned)(ns * 64),
                      16, LSTM_G, 128);
      } else {
        // Fallback: cooperative synchronous copy of next slice (visible after
        // the next stage barrier).
        const __bf16* src = Whh + (size_t)ns * 32;
        __bf16* dst = &wbuf[(s + 1) & 1][0];
        for (int r = tid; r < LSTM_G; r += 512) {
          u32x4 d0 = *(const u32x4*)(src + (size_t)r * LSTM_H);
          u32x4 d1 = *(const u32x4*)(src + (size_t)r * LSTM_H + 16);
          *(u32x4*)(dst + r * 32) = d0;
          *(u32x4*)(dst + r * 32 + 16) = d1;
        }
      }
      const __bf16* bb = &wbuf[s & 1][0];
#pragma unroll
      for (int gi = 0; gi < 4; ++gi) {
        const int n = gi * LSTM_H + wv * 16 + l15;
        FragA bf;
        const __bf16* bp = bb + n * 32 + hiK;
        bf.q[0] = *(const u32x4*)bp;
        bf.q[1] = *(const u32x4*)(bp + 16);
        acc[gi].v = __builtin_amdgcn_wmma_f32_16x16x32_bf16(
            false, ah[s].v, false, bf.v, (short)0, acc[gi].v, false, false);
      }
    }

    // ---- (D) LSTM cell update, fully in registers (wave owns i,f,g,o of j) ----
    float partial[8];
#pragma unroll
    for (int r = 0; r < 8; ++r) {
      const float ig = 1.f / (1.f + __expf(-acc[0].f[r]));
      const float fg = 1.f / (1.f + __expf(-acc[1].f[r]));
      const float gg = tanhf(acc[2].f[r]);
      const float og = 1.f / (1.f + __expf(-acc[3].f[r]));
      const float c  = fg * creg[r] + ig * gg;
      creg[r] = c;
      const float h = og * tanhf(c);
      hsh[(r + hiK) * LSTM_H + wv * 16 + l15] = (__bf16)h;  // feed layout for t+1
      partial[r] = h * wo;
    }
    // ---- (E) fused output projection: reduce over j within each 16-lane half
#pragma unroll
    for (int r = 0; r < 8; ++r) {
#pragma unroll
      for (int m = 8; m >= 1; m >>= 1) partial[r] += __shfl_xor(partial[r], m, 32);
    }
    if (l15 == 0) {  // lanes 0 (m=r) and 16 (m=r+8)
#pragma unroll
      for (int r = 0; r < 8; ++r) atomicAdd(&outacc[r + hiK], partial[r]);
    }
    __syncthreads();
    if (tid < 16) {
      Out[(size_t)(bm * 16 + tid) * LSTM_T + t] = outacc[tid] + bo;
      outacc[tid] = 0.0f;
    }
  }
}

// ---------------- launcher ----------------
extern "C" void kernel_launch(void* const* d_in, const int* in_sizes, int n_in,
                              void* d_out, int out_size, void* d_ws, size_t ws_size,
                              hipStream_t stream) {
  (void)in_sizes; (void)n_in; (void)out_size; (void)ws_size;
  const float* x    = (const float*)d_in[0];  // [64,2048,64]
  const float* W_ih = (const float*)d_in[1];  // [1024,64]
  const float* W_hh = (const float*)d_in[2];  // [1024,256]
  const float* b_ih = (const float*)d_in[3];  // [1024]
  const float* b_hh = (const float*)d_in[4];  // [1024]
  const float* W_o  = (const float*)d_in[5];  // [1,256]
  const float* b_o  = (const float*)d_in[6];  // [1]
  float* out = (float*)d_out;                 // [64,2048,1] f32

  char* ws = (char*)d_ws;
  __bf16* whh_bf = (__bf16*)(ws);             // 512 KB
  __bf16* wih_bf = (__bf16*)(ws + 524288);    // 128 KB
  float*  bsum   = (float*) (ws + 655360);    // 4 KB
  __bf16* x_bf   = (__bf16*)(ws + 659456);    // 16 MB

  const int nWhh = LSTM_G * LSTM_H;           // 262144
  const int nWih = LSTM_G * LSTM_I;           // 65536
  const int nX   = LSTM_B * LSTM_T * LSTM_I;  // 8388608

  cvt_bf16_kernel<<<(nWhh + 255) / 256, 256, 0, stream>>>(W_hh, whh_bf, nWhh);
  cvt_bf16_kernel<<<(nWih + 255) / 256, 256, 0, stream>>>(W_ih, wih_bf, nWih);
  cvt_bf16_kernel<<<(nX + 255) / 256, 256, 0, stream>>>(x, x_bf, nX);
  bias_sum_kernel<<<4, 256, 0, stream>>>(b_ih, b_hh, bsum);

  lstm_persist_kernel<<<4, 512, 0, stream>>>(whh_bf, wih_bf, x_bf, bsum,
                                             W_o, b_o, out);
}
